// Sg2ScVAEModel_81570018886298
// MI455X (gfx1250) — matrix operand order repaired and checked by reading, not compiled, not run on hardware
//
#include <hip/hip_runtime.h>
#include <cstdint>
#include <cstddef>

// ---------------------------------------------------------------------------
// Sg2Sc VAE forward for MI455X (gfx1250, wave32, WMMA + TDM).
// Activations and (pre-transposed) weights are bf16; GEMMs use
// v_wmma_f32_16x16x32_bf16 with f32 accumulate; tiles are staged into
// double-buffered LDS either by the Tensor Data Mover (tensor_load_to_lds,
// TENSORcnt-synchronized) or by vectorized copies as a fallback.
// ---------------------------------------------------------------------------

typedef __bf16 bf16;
typedef __attribute__((ext_vector_type(16))) __bf16 v16bf;
typedef __attribute__((ext_vector_type(8)))  __bf16 v8bf;
typedef __attribute__((ext_vector_type(8)))  float  v8f;

#define EMB   128
#define HID   512
#define NOBJ  4096
#define NEDGE 16384

#define BM 128
#define BN 128
#define BK 32
#define BPAD 8

// ---- Tensor Data Mover availability (device pass only; host pass falls back)
#if defined(__has_builtin)
#  if __has_builtin(__builtin_amdgcn_tensor_load_to_lds) && \
      __has_builtin(__builtin_amdgcn_s_wait_tensorcnt)
#    define HAVE_TDM 1
#  endif
#endif
#ifndef HAVE_TDM
#  define HAVE_TDM 0
#endif

#if HAVE_TDM
typedef __attribute__((ext_vector_type(4))) unsigned int v4u;
typedef __attribute__((ext_vector_type(8))) int          v8i;
typedef __attribute__((ext_vector_type(4))) int          v4i;

// Issue one 2-D tile load (global -> LDS) through the TDM.
// Dims/strides in data_size units (bf16 => data_size code 1 = 2 bytes).
// OOB rows/cols are zero-filled by the TDM (tensor_dim checks), so no guard
// code is needed in the GEMM staging path.
__device__ __forceinline__ void tdm_tile_load_bf16(
    unsigned lds_addr, const void* gsrc,
    unsigned tensor_d0, unsigned tensor_d1,
    unsigned tile_d0, unsigned tile_d1,
    unsigned long long stride0, unsigned padcfg)
{
    unsigned long long ga = (unsigned long long)gsrc;
    v4u g0;
    g0[0] = 1u;                                        // count=1, user desc
    g0[1] = lds_addr;                                  // LDS byte address
    g0[2] = (unsigned)(ga & 0xffffffffu);              // global_addr[31:0]
    g0[3] = (unsigned)((ga >> 32) & 0x01ffffffu)       // global_addr[56:32]
          | (2u << 30);                                // type=2 ("image")
    v8i g1;
    g1[0] = (int)((1u << 16) | padcfg);                // data_size=1 (2B) | pad cfg
    g1[1] = (int)((tensor_d0 & 0xffffu) << 16);        // tensor_dim0[15:0]
    g1[2] = (int)(((tensor_d0 >> 16) & 0xffffu) |
                  ((tensor_d1 & 0xffffu) << 16));      // dim0[31:16] | dim1[15:0]
    g1[3] = (int)(((tensor_d1 >> 16) & 0xffffu) |
                  ((tile_d0 & 0xffffu) << 16));        // dim1[31:16] | tile_dim0
    g1[4] = (int)(tile_d1 & 0xffffu);                  // tile_dim1 (tile_dim2=0)
    g1[5] = (int)(unsigned)(stride0 & 0xffffffffu);    // tensor_dim0_stride[31:0]
    g1[6] = (int)(unsigned)((stride0 >> 32) & 0xffffu);// stride0[47:32] (dim1_stride=0)
    g1[7] = 0;
    v4i g2 = (v4i)0;                                   // 2-D tensor: groups 2/3 zero
    v4i g3 = (v4i)0;
#if defined(__clang_major__) && __clang_major__ >= 23
    v8i gx = (v8i)0;
    __builtin_amdgcn_tensor_load_to_lds(g0, g1, g2, g3, gx, 0);
#else
    __builtin_amdgcn_tensor_load_to_lds(g0, g1, g2, g3, 0);
#endif
}
#endif // HAVE_TDM

// ---------------------------------------------------------------------------
// GEMM: C[M,N] = act(A[M,K](bf16,lda) @ B^T  + bias[N])
//   B is pre-transposed bf16: BT[N,K] with ldb (== K), so both operands stage
//   as row-tiles of contiguous K.
// Block: 256 threads = 8 waves; 128x128 tile; wave grid 2x4, each wave owns a
// 64x32 sub-tile = 4x2 WMMA 16x16x32 tiles with f32 accumulators.
// Double-buffered LDS; staging via TDM (preferred) or vector copies.
// ---------------------------------------------------------------------------
template <typename OutT, bool RELU>
__global__ __launch_bounds__(256, 2)
void gemm_bf16_wmma(const bf16* __restrict__ A, int lda,
                    const bf16* __restrict__ BT, int ldb,
                    const float* __restrict__ bias,
                    OutT* __restrict__ C, int ldc,
                    int M, int N, int K)
{
    __shared__ __align__(16) bf16 As[2][BM][BK];          // 2 x 8 KB
    __shared__ __align__(16) bf16 Bs[2][BN][BK + BPAD];   // 2 x 10 KB (padded rows)

    const int tid  = threadIdx.x;
    const int lane = tid & 31;
    const int wave = tid >> 5;
    const int wm   = (wave >> 2) * 64;   // 0 / 64
    const int wn   = (wave & 3) * 32;    // 0,32,64,96
    const int m0   = blockIdx.y * BM;
    const int n0   = blockIdx.x * BN;
    const int half = lane >> 4;          // 0/1
    const int l16  = lane & 15;

    v8f acc[4][2];
    for (int i = 0; i < 4; ++i)
        for (int j = 0; j < 2; ++j)
            for (int e = 0; e < 8; ++e) acc[i][j][e] = 0.0f;

#if HAVE_TDM
    // TDM staging: wave 0 issues two tile descriptors per stage (A and B).
    // B tile uses TDM LDS padding: every 16 DWORDs (64B row) insert 4 DWORDs
    // (16B = BPAD bf16) -> matches Bs[BN][BK+BPAD] layout.
    const unsigned padB = (1u << 20) | (3u << 22) | (3u << 25);
    auto issue = [&](int buf, int k0) {
        tdm_tile_load_bf16((unsigned)(size_t)&As[buf][0][0],
                           A + (size_t)m0 * lda + k0,
                           (unsigned)(K - k0), (unsigned)(M - m0),
                           BK, BM, (unsigned long long)lda, 0u);
        tdm_tile_load_bf16((unsigned)(size_t)&Bs[buf][0][0],
                           BT + (size_t)n0 * ldb + k0,
                           (unsigned)(K - k0), (unsigned)(N - n0),
                           BK, BN, (unsigned long long)ldb, padB);
    };
#else
    // Fallback: cooperative vectorized staging (16 bf16 per thread per tile).
    const int ar  = tid >> 1;            // tile row 0..127
    const int ac0 = (tid & 1) << 4;      // col 0 or 16
    auto stage = [&](int buf, int k0) {
        const int gm = m0 + ar;
        if (gm < M && (k0 + ac0 + 15) < K && !(lda & 7)) {
            const v8bf* s = (const v8bf*)(A + (size_t)gm * lda + k0 + ac0);
            *(v8bf*)&As[buf][ar][ac0]     = s[0];
            *(v8bf*)&As[buf][ar][ac0 + 8] = s[1];
        } else {
            for (int c = 0; c < 16; ++c) {
                int gk = k0 + ac0 + c;
                As[buf][ar][ac0 + c] =
                    (gm < M && gk < K) ? A[(size_t)gm * lda + gk] : (bf16)0.0f;
            }
        }
        const int gn = n0 + ar;
        if (gn < N && (k0 + ac0 + 15) < K && !(ldb & 7)) {
            const v8bf* s = (const v8bf*)(BT + (size_t)gn * ldb + k0 + ac0);
            *(v8bf*)&Bs[buf][ar][ac0]     = s[0];
            *(v8bf*)&Bs[buf][ar][ac0 + 8] = s[1];
        } else {
            for (int c = 0; c < 16; ++c) {
                int gk = k0 + ac0 + c;
                Bs[buf][ar][ac0 + c] =
                    (gn < N && gk < K) ? BT[(size_t)gn * ldb + gk] : (bf16)0.0f;
            }
        }
    };
#endif

    auto compute = [&](int buf) {
        // A 16x32 bf16 frag (ISA 7.12.2, interleaved): halves 0..7 -> K=8*half+c,
        // halves 8..15 -> K=16+8*half+c.  B 32x16 frag: same interleaved K order.
        v16bf a[4];
        #pragma unroll
        for (int i = 0; i < 4; ++i) {
            const bf16* ap = &As[buf][wm + i * 16 + l16][half * 8];
            v8bf lo = *(const v8bf*)ap;
            v8bf hi = *(const v8bf*)(ap + 16);
            #pragma unroll
            for (int e = 0; e < 8; ++e) { a[i][e] = lo[e]; a[i][8 + e] = hi[e]; }
        }
        v16bf b[2];
        #pragma unroll
        for (int j = 0; j < 2; ++j) {
            const bf16* bp = &Bs[buf][wn + j * 16 + l16][half * 8];
            v8bf lo = *(const v8bf*)bp;
            v8bf hi = *(const v8bf*)(bp + 16);
            #pragma unroll
            for (int e = 0; e < 8; ++e) { b[j][e] = lo[e]; b[j][8 + e] = hi[e]; }
        }
        #pragma unroll
        for (int i = 0; i < 4; ++i)
            #pragma unroll
            for (int j = 0; j < 2; ++j)
                acc[i][j] = __builtin_amdgcn_wmma_f32_16x16x32_bf16(
                    false, a[i], false, b[j], (short)0, acc[i][j], false, false);
    };

    const int nk = (K + BK - 1) / BK;
    int buf = 0;
#if HAVE_TDM
    if (wave == 0) issue(0, 0);
#else
    stage(0, 0);
#endif
    for (int kt = 0; kt < nk; ++kt) {
#if HAVE_TDM
        if (wave == 0) __builtin_amdgcn_s_wait_tensorcnt((short)0);
#endif
        __syncthreads();   // current buf staged; previous compute finished
        if (kt + 1 < nk) {
#if HAVE_TDM
            if (wave == 0) issue(buf ^ 1, (kt + 1) * BK);
#else
            stage(buf ^ 1, (kt + 1) * BK);
#endif
        }
        compute(buf);
        buf ^= 1;
    }

    // epilogue: C/D layout — VGPR e, lanes 0-15: M=e,N=lane; lanes 16-31: M=8+e
    for (int i = 0; i < 4; ++i) {
        for (int j = 0; j < 2; ++j) {
            int col = n0 + wn + j * 16 + l16;
            if (col >= N) continue;
            float bv = bias ? bias[col] : 0.0f;
            for (int e = 0; e < 8; ++e) {
                int row = m0 + wm + i * 16 + half * 8 + e;
                if (row >= M) continue;
                float v = acc[i][j][e] + bv;
                if (RELU) v = fmaxf(v, 0.0f);
                C[(size_t)row * ldc + col] = (OutT)v;
            }
        }
    }
}

// ---------------------------------------------------------------------------
// One-time weight prep: W[K,N] f32 -> WT[N,K] bf16 (tiled transpose in LDS)
// ---------------------------------------------------------------------------
__global__ void transpose_convert_kernel(const float* __restrict__ W,
                                         bf16* __restrict__ WT, int K, int N)
{
    __shared__ float tile[32][33];
    const int nb = blockIdx.x * 32, kb = blockIdx.y * 32;
    const int tx = threadIdx.x & 31, ty = threadIdx.x >> 5;   // 32 x 8
    for (int r = 0; r < 32; r += 8) {
        int k = kb + ty + r, n = nb + tx;
        tile[ty + r][tx] = (k < K && n < N) ? W[(size_t)k * N + n] : 0.0f;
    }
    __syncthreads();
    for (int r = 0; r < 32; r += 8) {
        int n = nb + ty + r, k = kb + tx;
        if (n < N && k < K) WT[(size_t)n * K + k] = (bf16)tile[tx][ty + r];
    }
}

// ---------------------------------------------------------------------------
// Graph plumbing kernels
// ---------------------------------------------------------------------------
__global__ void gather_edges_kernel(const int* __restrict__ edges,
                                    const bf16* __restrict__ ov, int d,
                                    const bf16* __restrict__ pv, int ldpv,
                                    bf16* __restrict__ t)
{
    int e = blockIdx.x;
    int s = edges[2 * e], o = edges[2 * e + 1];
    const uint2* ps = (const uint2*)(ov + (size_t)s * d);
    const uint2* pp = (const uint2*)(pv + (size_t)e * ldpv);
    const uint2* po = (const uint2*)(ov + (size_t)o * d);
    uint2* td = (uint2*)(t + (size_t)e * 3 * d);
    int n4 = d >> 2;
    for (int j = threadIdx.x; j < n4; j += blockDim.x) {
        td[j]          = ps[j];
        td[n4 + j]     = pp[j];
        td[2 * n4 + j] = po[j];
    }
}

__global__ void scatter_pool_kernel(const int* __restrict__ edges,
                                    const bf16* __restrict__ t2, int ldt2, int d,
                                    float* __restrict__ pooled)
{
    int e = blockIdx.x;
    int s = edges[2 * e], o = edges[2 * e + 1];
    const bf16* row = t2 + (size_t)e * ldt2;
    for (int j = threadIdx.x; j < HID; j += blockDim.x) {
        atomicAdd(&pooled[(size_t)s * HID + j], (float)row[j]);
        atomicAdd(&pooled[(size_t)o * HID + j], (float)row[HID + d + j]);
    }
}

__global__ void count_edges_kernel(const int* __restrict__ edges, float* __restrict__ cnt, int ne)
{
    int e = blockIdx.x * blockDim.x + threadIdx.x;
    if (e < ne) {
        atomicAdd(&cnt[edges[2 * e]], 1.0f);
        atomicAdd(&cnt[edges[2 * e + 1]], 1.0f);
    }
}

__global__ void zero_f32_kernel(float* __restrict__ p, size_t n)
{
    size_t i = (size_t)blockIdx.x * blockDim.x + threadIdx.x;
    if (i < n) p[i] = 0.0f;
}

__global__ void pool_div_kernel(const float* __restrict__ pooled,
                                const float* __restrict__ cnt,
                                bf16* __restrict__ out)
{
    int r = blockIdx.x;
    float inv = 1.0f / fmaxf(cnt[r], 1.0f);
    for (int j = threadIdx.x; j < HID; j += blockDim.x)
        out[(size_t)r * HID + j] = (bf16)(pooled[(size_t)r * HID + j] * inv);
}

__global__ void gather_rows_kernel(const float* __restrict__ tab,
                                   const int* __restrict__ idx,
                                   bf16* __restrict__ dst, int width, int ldd)
{
    int r = blockIdx.x;
    const float* src = tab + (size_t)(idx ? idx[r] : r) * width;
    bf16* d = dst + (size_t)r * ldd;
    for (int j = threadIdx.x; j < width; j += blockDim.x) d[j] = (bf16)src[j];
}

__global__ void copy_rows_bf16_kernel(const bf16* __restrict__ src, int lds_,
                                      bf16* __restrict__ dst, int ldd, int width)
{
    int r = blockIdx.x;
    const bf16* s = src + (size_t)r * lds_;
    bf16* d = dst + (size_t)r * ldd;
    for (int j = threadIdx.x; j < width; j += blockDim.x) d[j] = s[j];
}

__global__ void make_z_kernel(const float* __restrict__ mu, const float* __restrict__ lv,
                              const float* __restrict__ eps, bf16* __restrict__ z, int n)
{
    int i = blockIdx.x * blockDim.x + threadIdx.x;
    if (i < n) z[i] = (bf16)(mu[i] + eps[i] * __expf(0.5f * lv[i]));
}

// ---------------------------------------------------------------------------
// Host orchestration
// ---------------------------------------------------------------------------
static void launch_gemm(hipStream_t st, const bf16* A, int lda,
                        const bf16* BT, int ldb, const float* bias,
                        void* C, int ldc, int M, int N, int K,
                        bool relu, bool outF32)
{
    dim3 grid((N + BN - 1) / BN, (M + BM - 1) / BM);
    dim3 blk(256);
    if (outF32) {
        if (relu) gemm_bf16_wmma<float, true ><<<grid, blk, 0, st>>>(A, lda, BT, ldb, bias, (float*)C, ldc, M, N, K);
        else      gemm_bf16_wmma<float, false><<<grid, blk, 0, st>>>(A, lda, BT, ldb, bias, (float*)C, ldc, M, N, K);
    } else {
        if (relu) gemm_bf16_wmma<bf16, true ><<<grid, blk, 0, st>>>(A, lda, BT, ldb, bias, (bf16*)C, ldc, M, N, K);
        else      gemm_bf16_wmma<bf16, false><<<grid, blk, 0, st>>>(A, lda, BT, ldb, bias, (bf16*)C, ldc, M, N, K);
    }
}

struct Scratch {
    bf16 *t, *t2h, *t2, *poolbf, *h1;
    float *pooled, *cnt;
};

// one graph-conv layer; weight base index b: {n1a.W,b, n1b.W,b, n2a.W,b, n2b.W,b}
static void gconv_layer(hipStream_t st, const int* edges, bf16* ov, bf16* pv, int d,
                        int b, bf16* const* wt, void* const* d_in, Scratch& S)
{
    auto F = [&](int i) { return (const float*)d_in[7 + i]; };
    const int tw = 3 * d;
    const int ow = 2 * HID + d;
    gather_edges_kernel<<<NEDGE, 128, 0, st>>>(edges, ov, d, pv, d, S.t);
    launch_gemm(st, S.t,   tw,  wt[b + 0], tw,  F(b + 1), S.t2h, HID, NEDGE, HID, tw,  true, false);
    launch_gemm(st, S.t2h, HID, wt[b + 2], HID, F(b + 3), S.t2,  ow,  NEDGE, ow,  HID, true, false);
    copy_rows_bf16_kernel<<<NEDGE, 128, 0, st>>>(S.t2 + HID, ow, pv, d, d);   // new_p
    zero_f32_kernel<<<((size_t)NOBJ * HID + 255) / 256, 256, 0, st>>>(S.pooled, (size_t)NOBJ * HID);
    scatter_pool_kernel<<<NEDGE, 256, 0, st>>>(edges, S.t2, ow, d, S.pooled);
    pool_div_kernel<<<NOBJ, 256, 0, st>>>(S.pooled, S.cnt, S.poolbf);
    launch_gemm(st, S.poolbf, HID, wt[b + 4], HID, F(b + 5), S.h1, HID, NOBJ, HID, HID, true, false);
    launch_gemm(st, S.h1,     HID, wt[b + 6], HID, F(b + 7), ov,   d,   NOBJ, d,   HID, true, false);
}

static void gconv_net(hipStream_t st, const int* edges, bf16* ov, bf16* pv, int d,
                      int nlayers, int base, bf16* const* wt, void* const* d_in, Scratch& S)
{
    for (int l = 0; l < nlayers; ++l)
        gconv_layer(st, edges, ov, pv, d, base + l * 8, wt, d_in, S);
}

extern "C" void kernel_launch(void* const* d_in, const int* in_sizes, int n_in,
                              void* d_out, int out_size, void* d_ws, size_t ws_size,
                              hipStream_t stream)
{
    (void)in_sizes; (void)n_in; (void)out_size; (void)ws_size;

    // ---- inputs (setup_inputs() dict order) ----
    const int*   objs   = (const int*)d_in[0];
    const int*   edges  = (const int*)d_in[1];
    const int*   pidx   = (const int*)d_in[2];
    const float* boxes  = (const float*)d_in[3];
    const float* shapes = (const float*)d_in[4];
    const float* priors = (const float*)d_in[5];
    const float* eps    = (const float*)d_in[6];

    // params flattened in jax pytree order (dict keys sorted, W < b):
    // 0,1 box_emb | 2,3 box_mean | 4..7 box_mv | 8..11 box_net | 12,13 box_var
    // 14 emb_dc_box_obj | 15 emb_dc_box_pred | 16 emb_dc_shape_obj | 17 emb_dc_shape_pred
    // 18 emb_ec_box_obj | 19 emb_ec_box_pred | 20 emb_ec_shape_obj | 21 emb_ec_shape_pred
    // 22 gdc_box(40) | 62 gdc_shape(40) | 102 gec_box(40) | 142 gec_shape(40) | 182 gshared(24)
    // 206,207 shape_emb | 208,209 shape_mean | 210..213 shape_mv | 214..217 shape_net | 218,219 shape_var
    auto F = [&](int i) { return (const float*)d_in[7 + i]; };

    // ---- outputs ----
    float* out        = (float*)d_out;
    float* out_boxes  = out;                                  // O x 6
    float* out_shapes = out + (size_t)NOBJ * 6;               // O x 128
    float* out_mu     = out + (size_t)NOBJ * (6 + 128);       // O x 256
    float* out_lv     = out + (size_t)NOBJ * (6 + 128 + 256); // O x 256

    // ---- workspace bump allocator ----
    char* w = (char*)d_ws;
    auto alloc = [&](size_t bytes) -> void* {
        void* p = (void*)w;
        w += (bytes + 255) & ~(size_t)255;
        return p;
    };

    // ---- one-time weight prep: transpose + convert to bf16 (N x K) ----
    bf16* wt[220] = {};
    auto prepW = [&](int idx, int K, int N) {
        bf16* dst = (bf16*)alloc((size_t)K * N * sizeof(bf16));
        dim3 g((N + 31) / 32, (K + 31) / 32);
        transpose_convert_kernel<<<g, 256, 0, stream>>>(F(idx), dst, K, N);
        wt[idx] = dst;
    };
    prepW(0, 6, 128);                        // box_emb
    prepW(2, 256, 128);                      // box_mean
    prepW(4, 256, 512);  prepW(6, 512, 256); // box_mv
    prepW(8, 640, 512);  prepW(10, 512, 6);  // box_net
    prepW(12, 256, 128);                     // box_var
    auto prepNet = [&](int base, int nl, int d) {
        for (int l = 0; l < nl; ++l) {
            int b = base + l * 8;
            prepW(b + 0, 3 * d, HID);
            prepW(b + 2, HID, 2 * HID + d);
            prepW(b + 4, HID, HID);
            prepW(b + 6, HID, d);
        }
    };
    prepNet(22, 5, 640);   // gdc_box
    prepNet(62, 5, 640);   // gdc_shape
    prepNet(102, 5, 256);  // gec_box
    prepNet(142, 5, 256);  // gec_shape
    prepNet(182, 3, 512);  // gshared
    prepW(206, 128, 128);                        // shape_emb
    prepW(208, 256, 128);                        // shape_mean
    prepW(210, 256, 512);  prepW(212, 512, 256); // shape_mv
    prepW(214, 640, 512);  prepW(216, 512, 128); // shape_net
    prepW(218, 256, 128);                        // shape_var

    // ---- activation buffers ----
    bf16* boxes_bf  = (bf16*)alloc((size_t)NOBJ * 6 * 2);
    bf16* shapes_bf = (bf16*)alloc((size_t)NOBJ * 128 * 2);
    bf16* priors_bf = (bf16*)alloc((size_t)NOBJ * 128 * 2);
    bf16* ovB  = (bf16*)alloc((size_t)NOBJ * 256 * 2);
    bf16* ovS  = (bf16*)alloc((size_t)NOBJ * 256 * 2);
    bf16* ovSH = (bf16*)alloc((size_t)NOBJ * 512 * 2);
    bf16* dvB  = (bf16*)alloc((size_t)NOBJ * 640 * 2);
    bf16* dvS  = (bf16*)alloc((size_t)NOBJ * 640 * 2);
    bf16* zbuf = (bf16*)alloc((size_t)NOBJ * 256 * 2);
    bf16* h1   = (bf16*)alloc((size_t)NOBJ * 512 * 2);
    bf16* h2   = (bf16*)alloc((size_t)NOBJ * 256 * 2);
    bf16* pvA  = (bf16*)alloc((size_t)NEDGE * 640 * 2);
    bf16* pvB  = (bf16*)alloc((size_t)NEDGE * 640 * 2);
    bf16* pvC  = (bf16*)alloc((size_t)NEDGE * 512 * 2);
    Scratch S;
    S.t      = (bf16*)alloc((size_t)NEDGE * 1920 * 2);
    S.t2h    = (bf16*)alloc((size_t)NEDGE * 512 * 2);
    S.t2     = (bf16*)alloc((size_t)NEDGE * 1664 * 2);
    S.poolbf = (bf16*)alloc((size_t)NOBJ * HID * 2);
    S.h1     = h1;
    S.pooled = (float*)alloc((size_t)NOBJ * HID * 4);
    S.cnt    = (float*)alloc((size_t)NOBJ * 4);

    // ---- input conversions + degree counts ----
    gather_rows_kernel<<<NOBJ, 64, 0, stream>>>(boxes,  nullptr, boxes_bf,  6,   6);
    gather_rows_kernel<<<NOBJ, 128, 0, stream>>>(shapes, nullptr, shapes_bf, 128, 128);
    gather_rows_kernel<<<NOBJ, 128, 0, stream>>>(priors, nullptr, priors_bf, 128, 128);
    zero_f32_kernel<<<(NOBJ + 255) / 256, 256, 0, stream>>>(S.cnt, (size_t)NOBJ);
    count_edges_kernel<<<(NEDGE + 255) / 256, 256, 0, stream>>>(edges, S.cnt, NEDGE);

    // ================= encoder =================
    gather_rows_kernel<<<NOBJ, 128, 0, stream>>>(F(18), objs, ovB, 128, 256);
    launch_gemm(stream, boxes_bf, 6, wt[0], 6, F(1), ovB + 128, 256, NOBJ, 128, 6, false, false);
    gather_rows_kernel<<<NOBJ, 128, 0, stream>>>(F(20), objs, ovS, 128, 256);
    launch_gemm(stream, shapes_bf, 128, wt[206], 128, F(207), ovS + 128, 256, NOBJ, 128, 128, false, false);
    gather_rows_kernel<<<NEDGE, 128, 0, stream>>>(F(19), pidx, pvA, 256, 256);
    gather_rows_kernel<<<NEDGE, 128, 0, stream>>>(F(21), pidx, pvB, 256, 256);

    gconv_net(stream, edges, ovB, pvA, 256, 5, 102, wt, d_in, S);  // gec_box
    gconv_net(stream, edges, ovS, pvB, 256, 5, 142, wt, d_in, S);  // gec_shape

    copy_rows_bf16_kernel<<<NOBJ, 128, 0, stream>>>(ovB, 256, ovSH,       512, 256);
    copy_rows_bf16_kernel<<<NOBJ, 128, 0, stream>>>(ovS, 256, ovSH + 256, 512, 256);
    copy_rows_bf16_kernel<<<NEDGE, 128, 0, stream>>>(pvA, 256, pvC,       512, 256);
    copy_rows_bf16_kernel<<<NEDGE, 128, 0, stream>>>(pvB, 256, pvC + 256, 512, 256);
    gconv_net(stream, edges, ovSH, pvC, 512, 3, 182, wt, d_in, S); // gshared

    // heads: box branch (ov_sh cols 0..255)
    launch_gemm(stream, ovSH,       512, wt[4],   256, F(5),   h1, 512, NOBJ, 512, 256, true, false);
    launch_gemm(stream, h1,         512, wt[6],   512, F(7),   h2, 256, NOBJ, 256, 512, true, false);
    launch_gemm(stream, h2,         256, wt[2],   256, F(3),   out_mu,        256, NOBJ, 128, 256, false, true);
    launch_gemm(stream, h2,         256, wt[12],  256, F(13),  out_lv,        256, NOBJ, 128, 256, false, true);
    // heads: shape branch (ov_sh cols 256..511)
    launch_gemm(stream, ovSH + 256, 512, wt[210], 256, F(211), h1, 512, NOBJ, 512, 256, true, false);
    launch_gemm(stream, h1,         512, wt[212], 512, F(213), h2, 256, NOBJ, 256, 512, true, false);
    launch_gemm(stream, h2,         256, wt[208], 256, F(209), out_mu + 128,  256, NOBJ, 128, 256, false, true);
    launch_gemm(stream, h2,         256, wt[218], 256, F(219), out_lv + 128,  256, NOBJ, 128, 256, false, true);

    make_z_kernel<<<(NOBJ * 256 + 255) / 256, 256, 0, stream>>>(out_mu, out_lv, eps, zbuf, NOBJ * 256);

    // ================= decoder =================
    gather_rows_kernel<<<NOBJ, 128, 0, stream>>>(F(14), objs, dvB, 256, 640);
    copy_rows_bf16_kernel<<<NOBJ, 128, 0, stream>>>(priors_bf, 128, dvB + 256, 640, 128);
    copy_rows_bf16_kernel<<<NOBJ, 128, 0, stream>>>(zbuf,      256, dvB + 384, 640, 256);
    gather_rows_kernel<<<NOBJ, 128, 0, stream>>>(F(16), objs, dvS, 256, 640);
    copy_rows_bf16_kernel<<<NOBJ, 128, 0, stream>>>(priors_bf, 128, dvS + 256, 640, 128);
    copy_rows_bf16_kernel<<<NOBJ, 128, 0, stream>>>(zbuf,      256, dvS + 384, 640, 256);
    gather_rows_kernel<<<NEDGE, 128, 0, stream>>>(F(15), pidx, pvA, 640, 640);
    gather_rows_kernel<<<NEDGE, 128, 0, stream>>>(F(17), pidx, pvB, 640, 640);

    gconv_net(stream, edges, dvB, pvA, 640, 5, 22, wt, d_in, S);   // gdc_box
    gconv_net(stream, edges, dvS, pvB, 640, 5, 62, wt, d_in, S);   // gdc_shape

    launch_gemm(stream, dvB, 640, wt[8],   640, F(9),   h1, 512, NOBJ, 512, 640, true, false);
    launch_gemm(stream, h1,  512, wt[10],  512, F(11),  out_boxes,  6,   NOBJ, 6,   512, false, true);
    launch_gemm(stream, dvS, 640, wt[214], 640, F(215), h1, 512, NOBJ, 512, 640, true, false);
    launch_gemm(stream, h1,  512, wt[216], 512, F(217), out_shapes, 128, NOBJ, 128, 512, false, true);
}
